// MiniTriangularUpdate_76072460747364
// MI455X (gfx1250) — matrix operand order repaired and checked by reading, not compiled, not run on hardware
//
#include <hip/hip_runtime.h>
#include <hip/hip_bf16.h>

typedef unsigned short u16;
typedef unsigned int   u32;
typedef __attribute__((ext_vector_type(16))) __bf16 v16bf;
typedef __attribute__((ext_vector_type(8)))  float  v8f;
typedef int v4i32 __attribute__((__vector_size__(16)));

__device__ __forceinline__ u16 f32_to_bf16(float f) {
    union { float f; u32 u; } c; c.f = f;
    u32 r = c.u + 0x7FFFu + ((c.u >> 16) & 1u);   // round-to-nearest-even
    return (u16)(r >> 16);
}
__device__ __forceinline__ float bf16_to_f32(u16 h) {
    union { u32 u; float f; } c; c.u = ((u32)h) << 16;
    return c.f;
}
__device__ __forceinline__ float sigmoidf(float x) {
    return 1.0f / (1.0f + __expf(-x));
}

union FragBF { u16 u[16]; v16bf v; };

// ---- CDNA5 async global->LDS copy (guarded; falls back to sync staging) ----
#if defined(__HIP_DEVICE_COMPILE__) && __has_builtin(__builtin_amdgcn_global_load_async_to_lds_b128)
#define ASYNC_COPY 1
#endif

__device__ __forceinline__ void copy16_g2l(const u16* gp, u16* lp) {
#ifdef ASYNC_COPY
    __builtin_amdgcn_global_load_async_to_lds_b128(
        (__attribute__((address_space(1))) v4i32*)gp,
        (__attribute__((address_space(3))) v4i32*)lp, 0, 0);
#else
    *(uint4*)lp = *(const uint4*)gp;
#endif
}
__device__ __forceinline__ void wait_g2l() {
#ifdef ASYNC_COPY
#if __has_builtin(__builtin_amdgcn_s_wait_asynccnt)
    __builtin_amdgcn_s_wait_asynccnt(0);
#else
    asm volatile("s_wait_asynccnt 0x0" ::: "memory");
#endif
#endif
}

// ---------------------------------------------------------------------------
// Workspace layout (u16 elements):
//   planes : 4 * [4][32][256][256]  (a1, b1, a2T, b2T)        33,554,432
//   x12    : [4][256][256][64]                                 16,777,216
//   wbf_in : [2][128][128]  (pin, gin)                             32,768
//   wbf_out: [2][128][64]   (pout, gout)                           16,384
// ---------------------------------------------------------------------------
#define PLANE_GROUP_SZ   8388608u     // one group: 4*32*256*256
#define PLANE_SZ         65536u       // 256*256

// ---- kernel 0: weights f32 -> bf16 ----------------------------------------
__global__ void k0_cvt_weights(const float* __restrict__ wpin,
                               const float* __restrict__ wgin,
                               const float* __restrict__ wpout,
                               const float* __restrict__ wgout,
                               u16* __restrict__ wbf_in,
                               u16* __restrict__ wbf_out) {
    int i = blockIdx.x * 256 + threadIdx.x;
    if (i < 16384) {
        wbf_in[i]         = f32_to_bf16(wpin[i]);
        wbf_in[16384 + i] = f32_to_bf16(wgin[i]);
    }
    if (i < 8192) {
        wbf_out[i]        = f32_to_bf16(wpout[i]);
        wbf_out[8192 + i] = f32_to_bf16(wgout[i]);
    }
}

// ---- kernel 1: LN(128) + gated dual projection -> planar bf16 chunks ------
// block = 128 threads (4 waves). Each block = 16 rows (fixed b,r1; r2 span 16).
// Wave w produces channels [32w,32w+32) == one plane group.
// Groups 0,1 (a1,b1): LDS-staged, written as coalesced b128 runs along r2.
// Groups 2,3 (a2T,b2T): transposed planes, element-scattered (L2-resident).
__global__ void k1_ln_gated_proj(const float* __restrict__ x,
                                 const float* __restrict__ mask,
                                 const float* __restrict__ lnw,
                                 const float* __restrict__ lnb,
                                 const u16*  __restrict__ wbf,     // [2][128][128]
                                 u16*        __restrict__ planes) {
    const int tile = blockIdx.x;            // 16384 tiles
    const int b    = tile >> 12;
    const int r1   = (tile >> 4) & 255;
    const int r2b  = (tile & 15) << 4;
    const int t    = threadIdx.x;
    const int wave = t >> 5, lane = t & 31;

    __shared__ __align__(16) u16   hA[16][136];
    __shared__ __align__(16) u16   stg[2][2][16][16];   // waves 0,1 staging
    __shared__ float red[16][8][2];
    __shared__ float smask[16];

    const long rowg0 = ((long)(b * 256 + r1)) * 256 + r2b;

    // ---- LayerNorm phase: 8 threads per row, 16 channels each ----
    {
        const int r = t >> 3, q = t & 7;
        const float* xr = x + (rowg0 + r) * 128 + q * 16;
        float v[16], s = 0.f, ss = 0.f;
        #pragma unroll
        for (int c = 0; c < 16; ++c) { v[c] = xr[c]; s += v[c]; ss += v[c] * v[c]; }
        red[r][q][0] = s; red[r][q][1] = ss;
        if (q == 0) smask[r] = mask[rowg0 + r];
        __syncthreads();
        float S = 0.f, SS = 0.f;
        #pragma unroll
        for (int i = 0; i < 8; ++i) { S += red[r][i][0]; SS += red[r][i][1]; }
        const float m   = S * (1.0f / 128.0f);
        const float var = SS * (1.0f / 128.0f) - m * m;
        const float inv = rsqrtf(var + 1e-5f);
        #pragma unroll
        for (int c = 0; c < 16; ++c) {
            const int ch = q * 16 + c;
            hA[r][ch] = f32_to_bf16((v[c] - m) * inv * lnw[ch] + lnb[ch]);
        }
    }
    __syncthreads();

    // ---- WMMA phase ----
    const int Mrow   = lane & 15;
    const int laneHi = lane >> 4;
    const int n0     = wave * 32;

    for (int nt = 0; nt < 2; ++nt) {
        const int n = n0 + nt * 16 + Mrow;          // output channel
        v8f accp = {}; v8f accg = {};
        #pragma unroll
        for (int k0 = 0; k0 < 128; k0 += 32) {
            FragBF a, bp, bg;
            #pragma unroll
            for (int e = 0; e < 16; ++e) {          // A layout: contiguous k = 16*laneHi + e
                const int K = k0 + e + laneHi * 16;
                a.u[e] = hA[Mrow][K];
            }
            const u16* wp = wbf +         n * 128 + k0 + laneHi * 16;
            const u16* wg = wbf + 16384 + n * 128 + k0 + laneHi * 16;
            #pragma unroll
            for (int e = 0; e < 16; ++e) { bp.u[e] = wp[e]; bg.u[e] = wg[e]; }
            accp = __builtin_amdgcn_wmma_f32_16x16x32_bf16(false, a.v, false, bp.v,
                                                           (short)0, accp, false, false);
            accg = __builtin_amdgcn_wmma_f32_16x16x32_bf16(false, a.v, false, bg.v,
                                                           (short)0, accg, false, false);
        }
        const int grp = n >> 5;                     // == wave (uniform branch)
        const int d   = n & 31;
        if (grp < 2) {
            #pragma unroll
            for (int r = 0; r < 8; ++r) {
                const int M = r + laneHi * 8;       // C/D layout row
                stg[wave][nt][M][Mrow] =
                    f32_to_bf16(accp[r] * sigmoidf(accg[r]) * smask[M]);
            }
        } else {
            u16* plane = planes + (size_t)grp * PLANE_GROUP_SZ
                                + (size_t)(b * 32 + d) * PLANE_SZ;
            #pragma unroll
            for (int r = 0; r < 8; ++r) {
                const int M = r + laneHi * 8;
                plane[(size_t)(r2b + M) * 256 + r1] =
                    f32_to_bf16(accp[r] * sigmoidf(accg[r]) * smask[M]);
            }
        }
    }
    __syncthreads();

    // ---- coalesced writeback of channels 0..63 (16B runs along r2) ----
    {
        const int ch = t >> 1, half = t & 1;        // 64 channels x 2 halves
        const int w  = ch >> 5, nt = (ch >> 4) & 1, dcol = ch & 15;
        const int d  = ch & 31;
        union { u16 u[8]; uint4 q; } pk;
        #pragma unroll
        for (int m = 0; m < 8; ++m) pk.u[m] = stg[w][nt][half * 8 + m][dcol];
        u16* plane = planes + (size_t)(ch >> 5) * PLANE_GROUP_SZ
                            + (size_t)(b * 32 + d) * PLANE_SZ;
        *(uint4*)(plane + (size_t)r1 * 256 + r2b + half * 8) = pk.q;
    }
}

// ---- kernel 2: triangular einsums as per-plane 256x256x256 GEMMs ----------
// C[i,j] = sum_k A[i,k] * B[j,k]   (planes pre-transposed so both einsums match)
// block = 256 threads (8 waves) = one (plane, 64-wide j-strip).
// B strip (64x256 bf16) is async-copied into LDS once and shared by all waves;
// each wave computes two 16(i) x 64(j) strips from it.
__global__ void k2_triangle(const u16* __restrict__ planes,
                            u16*       __restrict__ x12) {
    const int blk  = blockIdx.x;                    // 1024 blocks
    const int pl   = blk >> 2;                      // 256 (b,dd) planes
    const int jt   = blk & 3;
    const int b    = pl >> 6;
    const int dd   = pl & 63;
    const int grpA = (dd < 32) ? 0 : 2;
    const int d    = dd & 31;

    const u16* Ap = planes + (size_t)grpA       * PLANE_GROUP_SZ
                           + (size_t)(b * 32 + d) * PLANE_SZ;
    const u16* Bp = planes + (size_t)(grpA + 1) * PLANE_GROUP_SZ
                           + (size_t)(b * 32 + d) * PLANE_SZ;

    const int j0   = jt * 64;
    const int t    = threadIdx.x;
    const int lane = t & 31, wave = t >> 5;

    __shared__ __align__(16) u16 Bbuf[64][272];     // padded: row stride 544B

    // ---- stage B strip into LDS (async DMA path) ----
    {
        const int jr  = t >> 2;                     // 64 rows
        const int seg = (t & 3) * 64;               // 4 x 64 u16 segments
        const u16* g = Bp + (size_t)(j0 + jr) * 256 + seg;
        u16* l = &Bbuf[jr][seg];
        #pragma unroll
        for (int c = 0; c < 8; ++c) copy16_g2l(g + c * 8, l + c * 8);
    }
    wait_g2l();
    __syncthreads();

    const int Mrow = lane & 15, laneHi = lane >> 4;

    #pragma unroll
    for (int half = 0; half < 2; ++half) {
        const int i0 = (wave + 8 * half) * 16;
        v8f acc[4] = {};
        const u16* arow = Ap + (size_t)(i0 + Mrow) * 256;
        for (int k0 = 0; k0 < 256; k0 += 32) {
            __builtin_prefetch(arow + k0 + 32, 0, 0);       // global_prefetch_b8
            FragBF a;
            #pragma unroll
            for (int e = 0; e < 16; ++e) {
                const int K = k0 + e + laneHi * 16;
                a.u[e] = arow[K];
            }
            #pragma unroll
            for (int jj = 0; jj < 4; ++jj) {                // A-fragment reuse x4
                FragBF bb;
                const u16* bl = &Bbuf[jj * 16 + Mrow][k0 + laneHi * 16];
                #pragma unroll
                for (int e = 0; e < 16; ++e) bb.u[e] = bl[e];
                acc[jj] = __builtin_amdgcn_wmma_f32_16x16x32_bf16(
                              false, a.v, false, bb.v, (short)0, acc[jj], false, false);
            }
        }
        #pragma unroll
        for (int jj = 0; jj < 4; ++jj) {
            #pragma unroll
            for (int r = 0; r < 8; ++r) {
                const int i = i0 + r + laneHi * 8;
                const int j = j0 + jj * 16 + Mrow;
                x12[(((size_t)(b * 256 + i)) * 256 + j) * 64 + dd] =
                    f32_to_bf16(acc[jj][r]);
            }
        }
    }
}

// ---- kernel 3: LN(64) + gated up-projection 64 -> 128, f32 output ---------
__global__ void k3_lnout_proj(const u16*  __restrict__ x12,
                              const float* __restrict__ lnw,
                              const float* __restrict__ lnb,
                              const u16*  __restrict__ wo,     // [2][128][64]
                              float*       __restrict__ out) {
    const int tile = blockIdx.x;             // 16384 tiles of 16 rows
    const int t    = threadIdx.x;            // 128
    const int wave = t >> 5, lane = t & 31;

    __shared__ __align__(16) u16 hA[16][72];
    __shared__ float red[16][8][2];

    const size_t row0 = (size_t)tile * 16;
    {
        const int r = t >> 3, q = t & 7;
        const u16* xr = x12 + (row0 + r) * 64 + q * 8;
        float v[8], s = 0.f, ss = 0.f;
        #pragma unroll
        for (int c = 0; c < 8; ++c) { v[c] = bf16_to_f32(xr[c]); s += v[c]; ss += v[c] * v[c]; }
        red[r][q][0] = s; red[r][q][1] = ss;
        __syncthreads();
        float S = 0.f, SS = 0.f;
        #pragma unroll
        for (int i = 0; i < 8; ++i) { S += red[r][i][0]; SS += red[r][i][1]; }
        const float m   = S * (1.0f / 64.0f);
        const float var = SS * (1.0f / 64.0f) - m * m;
        const float inv = rsqrtf(var + 1e-5f);
        #pragma unroll
        for (int c = 0; c < 8; ++c) {
            const int ch = q * 8 + c;
            hA[r][ch] = f32_to_bf16((v[c] - m) * inv * lnw[ch] + lnb[ch]);
        }
    }
    __syncthreads();

    const int Mrow = lane & 15, laneHi = lane >> 4;
    const int n0 = wave * 32;
    for (int nt = 0; nt < 2; ++nt) {
        const int n = n0 + nt * 16 + Mrow;
        v8f accp = {}; v8f accg = {};
        #pragma unroll
        for (int k0 = 0; k0 < 64; k0 += 32) {
            FragBF a, bp, bg;
            #pragma unroll
            for (int e = 0; e < 16; ++e) {
                const int K = k0 + e + laneHi * 16;
                a.u[e] = hA[Mrow][K];
            }
            const u16* wp = wo +        n * 64 + k0 + laneHi * 16;
            const u16* wg = wo + 8192 + n * 64 + k0 + laneHi * 16;
            #pragma unroll
            for (int e = 0; e < 16; ++e) { bp.u[e] = wp[e]; bg.u[e] = wg[e]; }
            accp = __builtin_amdgcn_wmma_f32_16x16x32_bf16(false, a.v, false, bp.v,
                                                           (short)0, accp, false, false);
            accg = __builtin_amdgcn_wmma_f32_16x16x32_bf16(false, a.v, false, bg.v,
                                                           (short)0, accg, false, false);
        }
        #pragma unroll
        for (int r = 0; r < 8; ++r) {
            const int M = r + laneHi * 8;
            out[(row0 + M) * 128 + n] = accp[r] * sigmoidf(accg[r]);
        }
    }
}

// ---------------------------------------------------------------------------
extern "C" void kernel_launch(void* const* d_in, const int* in_sizes, int n_in,
                              void* d_out, int out_size, void* d_ws, size_t ws_size,
                              hipStream_t stream) {
    const float* x        = (const float*)d_in[0];
    const float* mask     = (const float*)d_in[1];
    const float* ln_in_w  = (const float*)d_in[2];
    const float* ln_in_b  = (const float*)d_in[3];
    const float* w_pin    = (const float*)d_in[4];
    const float* w_gin    = (const float*)d_in[5];
    const float* ln_out_w = (const float*)d_in[6];
    const float* ln_out_b = (const float*)d_in[7];
    const float* w_pout   = (const float*)d_in[8];
    const float* w_gout   = (const float*)d_in[9];
    float* out = (float*)d_out;

    u16* planes  = (u16*)d_ws;                     // 33,554,432 u16
    u16* x12     = planes  + 33554432u;            // 16,777,216 u16
    u16* wbf_in  = x12     + 16777216u;            //     32,768 u16
    u16* wbf_out = wbf_in  +    32768u;            //     16,384 u16
    // total workspace: ~96.1 MiB

    k0_cvt_weights<<<64, 256, 0, stream>>>(w_pin, w_gin, w_pout, w_gout, wbf_in, wbf_out);
    k1_ln_gated_proj<<<16384, 128, 0, stream>>>(x, mask, ln_in_w, ln_in_b, wbf_in, planes);
    k2_triangle<<<1024, 256, 0, stream>>>(planes, x12);
    k3_lnout_proj<<<16384, 128, 0, stream>>>(x12, ln_out_w, ln_out_b, wbf_out, out);
}
